// InceptionTimeGAT_27582279975621
// MI455X (gfx1250) — matrix operand order repaired and checked by reading, not compiled, not run on hardware
//
#include <hip/hip_runtime.h>
#include <hip/hip_bf16.h>
#include <cstddef>

// 1/sqrt(1 + 1e-5)
#define BN_INV_C 0.9999950000375f

// flip to 0 if the async-to-LDS inline asm is rejected by the assembler
#define ASYNC_TILE_LOAD 1

typedef __attribute__((ext_vector_type(16))) __bf16 v16bf;
typedef __attribute__((ext_vector_type(8)))  __bf16 v8bf;
typedef __attribute__((ext_vector_type(8)))  float  v8f;

#define N_   512
#define SEQ_ 512
#define D_   128

// ---------------------------------------------------------------------------
// helpers
// ---------------------------------------------------------------------------
__device__ __forceinline__ v8bf bfmax8(v8bf a, v8bf b) {
    v8bf r;
#pragma unroll
    for (int i = 0; i < 8; ++i) {
        float fa = (float)a[i], fb = (float)b[i];
        r[i] = (__bf16)(fa > fb ? fa : fb);
    }
    return r;
}

// ---------------------------------------------------------------------------
// prep: x (N,12,SEQ) f32  ->  xbf (N,SEQ,32) bf16, channels 12..31 zero-padded
// ---------------------------------------------------------------------------
__global__ void pack_x_kernel(const float* __restrict__ x, __bf16* __restrict__ xbf) {
    int idx = blockIdx.x * blockDim.x + threadIdx.x;   // over N*SEQ
    if (idx >= N_ * SEQ_) return;
    int n = idx >> 9, s = idx & 511;
    size_t ob = (size_t)idx * 32;
#pragma unroll
    for (int c = 0; c < 32; ++c) {
        float v = (c < 12) ? x[((size_t)n * 12 + c) * SEQ_ + s] : 0.0f;
        xbf[ob + c] = (__bf16)v;
    }
}

// ---------------------------------------------------------------------------
// prep: pack conv weights W[Cout][Cin][k] f32 into WMMA A-fragment order bf16.
// flat index = (((cb*k + t)*numM + mtile)*32 + lane)*16 + e
// A layout used here (contiguous k): M = lane%16, K = 16*(lane/16) + e
// ---------------------------------------------------------------------------
__global__ void pack_wfrag_kernel(const float* __restrict__ W, __bf16* __restrict__ dst,
                                  int Cout, int Cin, int ksize, int numM, int total) {
    int idx = blockIdx.x * blockDim.x + threadIdx.x;
    if (idx >= total) return;
    int e     = idx & 15;
    int lane  = (idx >> 4) & 31;
    int rest  = idx >> 9;
    int mtile = rest % numM; rest /= numM;
    int t     = rest % ksize;
    int cb    = rest / ksize;
    int co    = mtile * 16 + (lane & 15);
    int ci    = cb * 32 + (lane >> 4) * 16 + e;
    float v = 0.0f;
    if (co < Cout && ci < Cin) v = W[((size_t)co * Cin + ci) * ksize + t];
    dst[idx] = (__bf16)v;
}

// ---------------------------------------------------------------------------
// implicit-GEMM conv (optionally fused maxpool3 on input, BN+ReLU on output)
// in  : bf16 (N, SEQ, CinPad) channel-last
// out : bf16 (N, SEQ, OutC), writes channels [co_off, co_off + 16*numM)
// per block: one (n, 128-wide seq tile); 8 waves; each wave owns one M-tile
// and a PAIR of 16-wide N-tiles so each A fragment feeds two WMMAs.
// Tile staging uses global_load_async_to_lds_b128 (ASYNCcnt) where possible.
// ---------------------------------------------------------------------------
__global__ __launch_bounds__(256) void conv_wmma_kernel(
        const __bf16* __restrict__ in, __bf16* __restrict__ out,
        const __bf16* __restrict__ wfrag,
        const float* __restrict__ bn_g, const float* __restrict__ bn_b,
        int CinPad, int CinBlocks, int ksize, int khalf,
        int numM, int co_off, int OutC, int doRelu, int doPool) {
    extern __shared__ __bf16 tile[];   // [128+k-1][CinPad]

    int n    = blockIdx.y;
    int s0   = blockIdx.x * 128;
    int tid  = threadIdx.x;
    int lane = tid & 31;
    int wave = tid >> 5;

    // ---- stage input tile (positions s0-khalf .. s0+127+khalf) into LDS ----
    int P = 128 + ksize - 1;
    int chunksPerRow = CinPad >> 3;                      // v8bf chunks per row
    int chShift = __builtin_ctz((unsigned)chunksPerRow); // CinPad power of two
    int chMask  = chunksPerRow - 1;
    for (int idx = tid; idx < P * chunksPerRow; idx += 256) {
        int p  = idx >> chShift;
        int ch = (idx & chMask) << 3;
        int s  = s0 - khalf + p;
        size_t ldsOff = (size_t)p * CinPad + ch;
        if (s >= 0 && s < SEQ_) {
            const __bf16* base = in + ((size_t)n * SEQ_ + s) * CinPad + ch;
            if (doPool) {   // maxpool3, SAME (identity = skip out-of-range)
                v8bf val = *(const v8bf*)base;
                if (s > 0)        val = bfmax8(val, *(const v8bf*)(base - CinPad));
                if (s < SEQ_ - 1) val = bfmax8(val, *(const v8bf*)(base + CinPad));
                *(v8bf*)(tile + ldsOff) = val;
            } else {
#if ASYNC_TILE_LOAD
                // dynamic LDS begins right after static LDS in the wave's space
                unsigned la = (unsigned)__builtin_amdgcn_groupstaticsize() +
                              (unsigned)(ldsOff * 2);
                asm volatile("global_load_async_to_lds_b128 %0, %1, off"
                             :: "v"(la), "v"(base) : "memory");
#else
                *(v8bf*)(tile + ldsOff) = *(const v8bf*)base;
#endif
            }
        } else {
            v8bf z;
#pragma unroll
            for (int i = 0; i < 8; ++i) z[i] = (__bf16)0.0f;
            *(v8bf*)(tile + ldsOff) = z;
        }
    }
#if ASYNC_TILE_LOAD
    asm volatile("s_wait_asynccnt 0" ::: "memory");
#endif
    __syncthreads();

    int group = lane >> 4;
    int lr    = lane & 15;
    int jobs  = numM * 4;                 // numM x (4 pairs of N-tiles)
    size_t astep = (size_t)numM * 512;    // A-fragment stride per tap t (elems)
    for (int job = wave; job < jobs; job += 8) {
        int mtile = job >> 2;
        int pr    = job & 3;              // pair index -> ntiles {2pr, 2pr+1}
        v8f acc0 = {0.f, 0.f, 0.f, 0.f, 0.f, 0.f, 0.f, 0.f};
        v8f acc1 = {0.f, 0.f, 0.f, 0.f, 0.f, 0.f, 0.f, 0.f};

        for (int cb = 0; cb < CinBlocks; ++cb) {
            const __bf16* aptr = wfrag +
                (((size_t)cb * ksize * numM + mtile) * 32 + lane) * 16;
            // B: K = channels cb*32 + group*16 + e, N = position base + lr + t
            const __bf16* tb0 =
                tile + (size_t)(pr * 32 + lr) * CinPad + cb * 32 + group * 16;
            const __bf16* tb1 = tb0 + (size_t)16 * CinPad;
            for (int t = 0; t < ksize; ++t) {
                v16bf A  = *(const v16bf*)aptr;
                __builtin_prefetch(aptr + astep, 0, 1);  // next tap's fragment
                v16bf B0 = *(const v16bf*)tb0;
                v16bf B1 = *(const v16bf*)tb1;
                acc0 = __builtin_amdgcn_wmma_f32_16x16x32_bf16(
                           false, A, false, B0, (short)0, acc0, false, false);
                acc1 = __builtin_amdgcn_wmma_f32_16x16x32_bf16(
                           false, A, false, B1, (short)0, acc1, false, false);
                aptr += astep;
                tb0  += CinPad;
                tb1  += CinPad;
            }
        }

        // C/D layout: VGPR r -> M = r + 8*group (lanes 0-15 / 16-31), N = lr
#pragma unroll
        for (int which = 0; which < 2; ++which) {
            v8f acc = which ? acc1 : acc0;
            int ntile = pr * 2 + which;
            int s  = s0 + ntile * 16 + lr;
            int c0 = mtile * 16 + group * 8;
            size_t oidx = ((size_t)n * SEQ_ + s) * OutC + co_off + c0;
            v8bf ov;
#pragma unroll
            for (int r = 0; r < 8; ++r) {
                float v = acc[r];
                if (bn_g) { int c = c0 + r; v = v * (bn_g[c] * BN_INV_C) + bn_b[c]; }
                if (doRelu) v = v > 0.f ? v : 0.f;
                ov[r] = (__bf16)v;
            }
            *(v8bf*)(out + oidx) = ov;   // 16B packed store
        }
    }
}

// ---------------------------------------------------------------------------
// h = relu(h + bn(res))   (res raw conv output for d==2, activation for d==5)
// ---------------------------------------------------------------------------
__global__ void add_residual_kernel(__bf16* __restrict__ h, const __bf16* __restrict__ res,
                                    const float* __restrict__ g, const float* __restrict__ b,
                                    int total) {
    int idx = blockIdx.x * blockDim.x + threadIdx.x;
    if (idx >= total) return;
    int c = idx & 127;
    float v = (float)h[idx] + (float)res[idx] * (g[c] * BN_INV_C) + b[c];
    h[idx] = (__bf16)(v > 0.f ? v : 0.f);
}

// ---------------------------------------------------------------------------
// hidden[n][c] = mean over s of h[n][s][c]
// ---------------------------------------------------------------------------
__global__ void mean_seq_kernel(const __bf16* __restrict__ h, float* __restrict__ hidden) {
    int n = blockIdx.x, c = threadIdx.x;   // 128 threads
    float acc = 0.f;
    for (int s = 0; s < SEQ_; ++s)
        acc += (float)h[((size_t)n * SEQ_ + s) * D_ + c];
    hidden[n * D_ + c] = acc * (1.0f / SEQ_);
}

// ---------------------------------------------------------------------------
// out[r][m] = act( in[r][:] . W[m][:] + bias[m] ),  K = 128 fixed
// ---------------------------------------------------------------------------
__global__ void linear_kernel(const float* __restrict__ in, const float* __restrict__ W,
                              const float* __restrict__ bias, float* __restrict__ out,
                              int M, int act) {
    __shared__ float row[128];
    int r = blockIdx.x, tid = threadIdx.x;   // 128 threads
    row[tid] = in[(size_t)r * 128 + tid];
    __syncthreads();
    for (int m = tid; m < M; m += 128) {
        float acc = bias[m];
        for (int k = 0; k < 128; ++k) acc += row[k] * W[(size_t)m * 128 + k];
        if (act) acc = acc >= 0.f ? acc : 0.01f * acc;
        out[(size_t)r * M + m] = acc;
    }
}

// ---------------------------------------------------------------------------
// s[i] = x1[i].a[0:128], t[i] = x1[i].a[128:256]
// ---------------------------------------------------------------------------
__global__ void st_kernel(const float* __restrict__ x1, const float* __restrict__ a,
                          float* __restrict__ sv, float* __restrict__ tv) {
    int i = blockIdx.x * blockDim.x + threadIdx.x;
    if (i >= N_) return;
    float s = 0.f, t = 0.f;
    for (int k = 0; k < 128; ++k) {
        float v = x1[(size_t)i * 128 + k];
        s += v * a[k];
        t += v * a[128 + k];
    }
    sv[i] = s;
    tv[i] = t;
}

// ---------------------------------------------------------------------------
// row-wise softmax of leaky(t_i + s_j) then hidden2 = att@hidden + hidden
// one block per row i; 128 threads.
// ---------------------------------------------------------------------------
__global__ void att_kernel(const float* __restrict__ sv, const float* __restrict__ tv,
                           const float* __restrict__ hidden,
                           float* __restrict__ hid2, float* __restrict__ outH2) {
    __shared__ float ss[512];
    __shared__ float pp[512];
    __shared__ float red[128];
    int i = blockIdx.x, tid = threadIdx.x;
    for (int j = tid; j < 512; j += 128) ss[j] = sv[j];
    __syncthreads();
    float ti = tv[i];
    float mx = -3.4e38f;
    for (int j = tid; j < 512; j += 128) {
        float sc = ti + ss[j];
        sc = sc >= 0.f ? sc : 0.01f * sc;
        mx = fmaxf(mx, sc);
    }
    red[tid] = mx; __syncthreads();
    for (int o = 64; o > 0; o >>= 1) {
        if (tid < o) red[tid] = fmaxf(red[tid], red[tid + o]);
        __syncthreads();
    }
    mx = red[0]; __syncthreads();
    float sum = 0.f;
    for (int j = tid; j < 512; j += 128) {
        float sc = ti + ss[j];
        sc = sc >= 0.f ? sc : 0.01f * sc;
        float p = __expf(sc - mx);
        pp[j] = p;
        sum += p;
    }
    red[tid] = sum; __syncthreads();
    for (int o = 64; o > 0; o >>= 1) {
        if (tid < o) red[tid] += red[tid + o];
        __syncthreads();
    }
    float inv = 1.f / red[0];
    float acc = 0.f;
    for (int j = 0; j < 512; ++j) acc += pp[j] * hidden[(size_t)j * 128 + tid];
    float v = acc * inv + hidden[(size_t)i * 128 + tid];
    hid2[(size_t)i * 128 + tid]  = v;
    outH2[(size_t)i * 128 + tid] = v;
}

// ---------------------------------------------------------------------------
// host driver
// ---------------------------------------------------------------------------
extern "C" void kernel_launch(void* const* d_in, const int* in_sizes, int n_in,
                              void* d_out, int out_size, void* d_ws, size_t ws_size,
                              hipStream_t stream) {
    (void)in_sizes; (void)n_in; (void)out_size; (void)ws_size;

    char* ws = (char*)d_ws;
    size_t off = 0;
    auto alloc = [&](size_t bytes) -> char* {
        char* p = ws + off;
        off += (bytes + 255) & ~(size_t)255;
        return p;
    };

    const size_t actSmall = (size_t)N_ * SEQ_ * 32 * 2;   // 32-ch bf16
    const size_t actBig   = (size_t)N_ * SEQ_ * 128 * 2;  // 128-ch bf16

    __bf16* XBF  = (__bf16*)alloc(actSmall);
    __bf16* H1   = (__bf16*)alloc(actBig);
    __bf16* H2   = (__bf16*)alloc(actBig);
    __bf16* H3   = (__bf16*)alloc(actBig);
    __bf16* BOTT = (__bf16*)alloc(actSmall);
    float*  HIDDEN = (float*)alloc((size_t)N_ * 128 * 4);
    float*  X1     = (float*)alloc((size_t)N_ * 128 * 4);
    float*  SV     = (float*)alloc((size_t)N_ * 4);
    float*  TV     = (float*)alloc((size_t)N_ * 4);
    float*  HID2   = (float*)alloc((size_t)N_ * 128 * 4);
    float*  GBUF   = (float*)alloc((size_t)N_ * 128 * 4);

    // ---- pack weights into WMMA fragment layouts ----
    auto packConv = [&](const float* W, int Cout, int Cin, int ksize,
                        int CB, int numM) -> const __bf16* {
        size_t elems = (size_t)CB * ksize * numM * 512;
        __bf16* dst = (__bf16*)alloc(elems * 2);
        int total = (int)elems;
        pack_wfrag_kernel<<<(total + 255) / 256, 256, 0, stream>>>(
            W, dst, Cout, Cin, ksize, numM, total);
        return dst;
    };

    pack_x_kernel<<<(N_ * SEQ_ + 255) / 256, 256, 0, stream>>>((const float*)d_in[0], XBF);

    const int ks[3] = {39, 19, 9};
    const __bf16 *bottF[6], *convF[6][3], *mpF[6];
    for (int d = 0; d < 6; ++d) {
        int ni = (d == 0) ? 12 : 128;
        int CB = (d == 0) ? 1 : 4;
        bottF[d] = packConv((const float*)d_in[1 + 7 * d], 32, ni, 1, CB, 2);
        for (int j = 0; j < 3; ++j)
            convF[d][j] = packConv((const float*)d_in[2 + 7 * d + j], 32, 32, ks[j], 1, 2);
        mpF[d] = packConv((const float*)d_in[5 + 7 * d], 32, ni, 1, CB, 2);
    }
    const __bf16* scF = packConv((const float*)d_in[43], 128, 12, 1, 1, 8);

    // ---- inception modules ----
    struct IO { const __bf16* in; __bf16* out; };
    IO io[6] = { {XBF, H1}, {H1, H2}, {H2, H3}, {H3, H1}, {H1, H2}, {H2, H1} };

    dim3 grid(SEQ_ / 128, N_), blk(256);
    int totalBig = N_ * SEQ_ * 128;

    for (int d = 0; d < 6; ++d) {
        const __bf16* IN = io[d].in;
        __bf16* OUT = io[d].out;
        int CinPad = (d == 0) ? 32 : 128;
        int CB     = (d == 0) ? 1 : 4;
        const float* bng = (const float*)d_in[6 + 7 * d];
        const float* bnb = (const float*)d_in[7 + 7 * d];

        // bottleneck 1x1 (no bn/relu)
        conv_wmma_kernel<<<grid, blk, 128 * CinPad * 2, stream>>>(
            IN, BOTT, bottF[d], nullptr, nullptr,
            CinPad, CB, 1, 0, 2, 0, 32, 0, 0);

        // three conv branches (bn+relu fused)
        for (int j = 0; j < 3; ++j) {
            int k = ks[j];
            conv_wmma_kernel<<<grid, blk, (128 + k - 1) * 32 * 2, stream>>>(
                BOTT, OUT, convF[d][j], bng + 32 * j, bnb + 32 * j,
                32, 1, k, k / 2, 2, 32 * j, 128, 1, 0);
        }

        // maxpool3 + 1x1 conv branch (bn+relu fused)
        conv_wmma_kernel<<<grid, blk, 128 * CinPad * 2, stream>>>(
            IN, OUT, mpF[d], bng + 96, bnb + 96,
            CinPad, CB, 1, 0, 2, 96, 128, 1, 1);

        if (d == 2) {
            // sc = bn(conv1x1(x)); h = relu(h + sc); H1 is free -> temp
            conv_wmma_kernel<<<grid, blk, 128 * 32 * 2, stream>>>(
                XBF, H1, scF, nullptr, nullptr,
                32, 1, 1, 0, 8, 0, 128, 0, 0);
            add_residual_kernel<<<(totalBig + 255) / 256, 256, 0, stream>>>(
                H3, H1, (const float*)d_in[44], (const float*)d_in[45], totalBig);
        } else if (d == 5) {
            // sc = bn(res=H3); h(=H1) = relu(h + sc)
            add_residual_kernel<<<(totalBig + 255) / 256, 256, 0, stream>>>(
                H1, H3, (const float*)d_in[46], (const float*)d_in[47], totalBig);
        }
    }

    // ---- head ----
    float* out_origin = (float*)d_out;                      // (512,7)
    float* out_gat    = (float*)d_out + (size_t)N_ * 7;     // (512,7)
    float* out_h2     = (float*)d_out + (size_t)2 * N_ * 7; // (512,128)

    mean_seq_kernel<<<N_, 128, 0, stream>>>(H1, HIDDEN);

    linear_kernel<<<N_, 128, 0, stream>>>(HIDDEN, (const float*)d_in[51],
                                          (const float*)d_in[52], out_origin, 7, 0);
    linear_kernel<<<N_, 128, 0, stream>>>(HIDDEN, (const float*)d_in[48],
                                          (const float*)d_in[49], X1, 128, 0);
    st_kernel<<<(N_ + 255) / 256, 256, 0, stream>>>(X1, (const float*)d_in[50], SV, TV);
    att_kernel<<<N_, 128, 0, stream>>>(SV, TV, HIDDEN, HID2, out_h2);
    linear_kernel<<<N_, 128, 0, stream>>>(HID2, (const float*)d_in[53],
                                          (const float*)d_in[54], GBUF, 128, 1);
    linear_kernel<<<N_, 128, 0, stream>>>(GBUF, (const float*)d_in[55],
                                          (const float*)d_in[56], out_gat, 7, 0);
}
